// DCGRUCell_61718680043778
// MI455X (gfx1250) — matrix operand order repaired and checked by reading, not compiled, not run on hardware
//
#include <hip/hip_runtime.h>

typedef __attribute__((ext_vector_type(16))) __bf16 v16bf;
typedef __attribute__((ext_vector_type(8)))  __bf16 v8bf;
typedef __attribute__((ext_vector_type(8)))  float  v8f;

#define NB    2048            // num nodes
#define BB    64              // batch
#define FD    128             // concat feature dim (input 64 + units 64)
#define UD    64              // rnn units (== input dim)
#define COLS  (BB*FD)         // 8192, diffusion GEMM column count
#define RR    (NB*BB)         // 131072, projection GEMM row count
#define MMAT  5               // NUM_MATRICES

// ---------------------------------------------------------------------------
// Support construction: S0 = (D^-1 A)^T, S1 = (D'^-1 A^T)^T, stored bf16
// ---------------------------------------------------------------------------
__global__ void k_degree(const float* __restrict__ A,
                         float* __restrict__ invrow, float* __restrict__ invcol) {
  __shared__ float sr[256], sc[256];
  const int j = blockIdx.x;
  float r = 0.f, c = 0.f;
  for (int k = threadIdx.x; k < NB; k += 256) {
    r += A[(size_t)j * NB + k];     // row sum of A
    c += A[(size_t)k * NB + j];     // col sum of A (= row sum of A^T)
  }
  sr[threadIdx.x] = r; sc[threadIdx.x] = c;
  __syncthreads();
  for (int s = 128; s > 0; s >>= 1) {
    if ((int)threadIdx.x < s) {
      sr[threadIdx.x] += sr[threadIdx.x + s];
      sc[threadIdx.x] += sc[threadIdx.x + s];
    }
    __syncthreads();
  }
  if (threadIdx.x == 0) {
    invrow[j] = sr[0] > 0.f ? 1.f / sr[0] : 0.f;
    invcol[j] = sc[0] > 0.f ? 1.f / sc[0] : 0.f;
  }
}

__global__ void k_supports(const float* __restrict__ A,
                           const float* __restrict__ invrow,
                           const float* __restrict__ invcol,
                           __bf16* __restrict__ S0, __bf16* __restrict__ S1) {
  size_t idx = (size_t)blockIdx.x * blockDim.x + threadIdx.x;   // over NB*NB
  int i = (int)(idx / NB), j = (int)(idx % NB);
  S0[idx] = (__bf16)(A[(size_t)j * NB + i] * invrow[j]);  // (D^-1 A)^T
  S1[idx] = (__bf16)(A[idx] * invcol[j]);                 // (D'^-1 A^T)^T
}

// ---------------------------------------------------------------------------
// Misc elementwise kernels
// ---------------------------------------------------------------------------
__global__ void k_cast_bf16(const float* __restrict__ src, __bf16* __restrict__ dst, int n) {
  int i = blockIdx.x * blockDim.x + threadIdx.x;
  if (i < n) dst[i] = (__bf16)src[i];
}

// X0[n][b][f] = f<UD ? inputs[b][n][f] : hx[b][n][f-UD]
__global__ void k_concat(const float* __restrict__ xi, const float* __restrict__ hx,
                         __bf16* __restrict__ X0) {
  size_t idx = (size_t)blockIdx.x * blockDim.x + threadIdx.x;   // over NB*BB*FD
  int f = (int)(idx % FD);
  size_t nb = idx / FD;
  int b = (int)(nb % BB);
  int n = (int)(nb / BB);
  float v = (f < UD) ? xi[(size_t)b * (NB * UD) + (size_t)n * UD + f]
                     : hx[(size_t)b * (NB * UD) + (size_t)n * UD + (f - UD)];
  X0[idx] = (__bf16)v;
}

__global__ void k_init_acc(float* __restrict__ ACC, const float* __restrict__ bias, int P) {
  size_t idx = (size_t)blockIdx.x * blockDim.x + threadIdx.x;   // over RR*P
  ACC[idx] = bias[idx % P];
}

// ---------------------------------------------------------------------------
// bf16 WMMA GEMM. Block = 128 threads = 4 waves (2 wave-rows x 2 wave-cols).
// Per wave: 2 row-tiles x CT col-tiles of 16x16 (C tile 32 x CT*16).
// Block tile: 64 rows x CT*32 cols.  K-step 32.
//   mode 0: Yb = A@B (bf16)
//   mode 1: Yb = 2*(A@B) - Z (bf16, Chebyshev step)
//   mode 2: Yf += A@B (f32 accumulate)
// A row-major [M x K] stride lda; B row-major [K x cols] stride ldb.
// ---------------------------------------------------------------------------
template<int CT>
__global__ __launch_bounds__(128)
void k_wmma_gemm(const __bf16* __restrict__ A, int lda,
                 const __bf16* __restrict__ Bm, int ldb,
                 const __bf16* __restrict__ Z,
                 __bf16* __restrict__ Yb, float* __restrict__ Yf,
                 int Kdim, int ldc, int mode)
{
  constexpr int BCOLS = CT * 32;     // block column width (64 or 128)
  constexpr int CPR   = BCOLS / 8;   // 16-byte chunks per staged B row
  __shared__ __attribute__((aligned(16))) __bf16 Bs[BCOLS][40];  // [col][k], padded

  const int tid   = threadIdx.x;
  const int lane  = tid & 31;
  const int wave  = tid >> 5;
  const int wr    = wave >> 1, wc = wave & 1;
  const int rowBlock = blockIdx.y * 64;
  const int colBlock = blockIdx.x * BCOLS;
  const int lhalf = lane >> 4;   // 0..1
  const int l15   = lane & 15;

  v8f acc[2][CT] = {};

  for (int k0 = 0; k0 < Kdim; k0 += 32) {
    // Stage B tile transposed: vector global loads (row direction), scalar LDS
    // scatter.  Bs[n][k] = B[k0+k][colBlock+n]
    for (int c = tid; c < 32 * CPR; c += 128) {
      int kk  = c / CPR;           // 0..31   (CPR is a power of two)
      int nn0 = (c % CPR) * 8;     // 0..BCOLS-8
      const __bf16* gp = Bm + (size_t)(k0 + kk) * ldb + colBlock + nn0;
      __builtin_prefetch(gp + 32 * (size_t)ldb, 0, 3);   // next K slab
      v8bf v = *reinterpret_cast<const v8bf*>(gp);
#pragma unroll
      for (int j = 0; j < 8; ++j) Bs[nn0 + j][kk] = v[j];
    }
    __syncthreads();

    // A fragments: 16-bit A 16x32 layout — lanes 0-15 K half 0, lanes 16-31 half 1;
    // vector elems [0..7] = K(khalf*8 + 0..7), [8..15] = K(16 + khalf*8 + 0..7)
    v16bf af[2];
#pragma unroll
    for (int tr = 0; tr < 2; ++tr) {
      const __bf16* ap = A + (size_t)(rowBlock + wr * 32 + tr * 16 + l15) * lda
                           + k0 + lhalf * 8;
      __builtin_prefetch(ap + 32, 0, 3);   // next K slab
      v8bf lo = *reinterpret_cast<const v8bf*>(ap);
      v8bf hi = *reinterpret_cast<const v8bf*>(ap + 16);
#pragma unroll
      for (int j = 0; j < 8; ++j) { af[tr][j] = lo[j]; af[tr][8 + j] = hi[j]; }
    }

    // B fragments: 16-bit B 32x16 layout — lane n = l15, K = lhalf*8 + j (j<8), 16 + lhalf*8 + (j-8) (j>=8)
    v16bf bfr[CT];
#pragma unroll
    for (int tc = 0; tc < CT; ++tc) {
      const __bf16* bp = &Bs[wc * (CT * 16) + tc * 16 + l15][lhalf * 8];
      v8bf lo = *reinterpret_cast<const v8bf*>(bp);
      v8bf hi = *reinterpret_cast<const v8bf*>(bp + 16);
#pragma unroll
      for (int j = 0; j < 8; ++j) { bfr[tc][j] = lo[j]; bfr[tc][8 + j] = hi[j]; }
    }

#pragma unroll
    for (int tr = 0; tr < 2; ++tr)
#pragma unroll
      for (int tc = 0; tc < CT; ++tc)
        acc[tr][tc] = __builtin_amdgcn_wmma_f32_16x16x32_bf16(
            false, af[tr], false, bfr[tc], (short)0, acc[tr][tc], false, false);

    __syncthreads();
  }

  // Write back: C/D 16x16 f32 layout — row = lhalf*8 + i, col = l15
#pragma unroll
  for (int tr = 0; tr < 2; ++tr) {
    int row0 = rowBlock + wr * 32 + tr * 16 + lhalf * 8;
#pragma unroll
    for (int tc = 0; tc < CT; ++tc) {
      int col = colBlock + wc * (CT * 16) + tc * 16 + l15;
#pragma unroll
      for (int i = 0; i < 8; ++i) {
        size_t idx = (size_t)(row0 + i) * ldc + col;
        float v = acc[tr][tc][i];
        if (mode == 2)      Yf[idx] += v;
        else if (mode == 1) Yb[idx] = (__bf16)(2.f * v - (float)Z[idx]);
        else                Yb[idx] = (__bf16)v;
      }
    }
  }
}

// ---------------------------------------------------------------------------
// Gate activation: r,u = sigmoid(ACCru); store u back; hidden half of X0 := r*h
// ---------------------------------------------------------------------------
__global__ void k_act_gate(float* __restrict__ ACC, const float* __restrict__ hx,
                           __bf16* __restrict__ X0) {
  size_t idx = (size_t)blockIdx.x * blockDim.x + threadIdx.x;   // over RR*UD
  int o = (int)(idx % UD);
  size_t r = idx / UD;              // r = n*BB + b
  int b = (int)(r % BB);
  int n = (int)(r / BB);
  float rg = 1.f / (1.f + expf(-ACC[r * (size_t)FD + o]));
  float ug = 1.f / (1.f + expf(-ACC[r * (size_t)FD + UD + o]));
  ACC[r * (size_t)FD + UD + o] = ug;   // keep u for the final blend
  float h = hx[(size_t)b * (NB * UD) + (size_t)n * UD + o];
  X0[r * (size_t)FD + UD + o] = (__bf16)(rg * h);
}

// new_state = u*h + (1-u)*tanh(c)
__global__ void k_final(const float* __restrict__ ACCru, const float* __restrict__ ACCc,
                        const float* __restrict__ hx, float* __restrict__ out) {
  size_t idx = (size_t)blockIdx.x * blockDim.x + threadIdx.x;   // over RR*UD
  int o = (int)(idx % UD);
  size_t r = idx / UD;
  int b = (int)(r % BB);
  int n = (int)(r / BB);
  float u = ACCru[r * (size_t)FD + UD + o];
  float c = tanhf(ACCc[r * (size_t)UD + o]);
  float h = hx[(size_t)b * (NB * UD) + (size_t)n * UD + o];
  out[(size_t)b * (NB * UD) + (size_t)n * UD + o] = u * h + (1.f - u) * c;
}

// ---------------------------------------------------------------------------
extern "C" void kernel_launch(void* const* d_in, const int* in_sizes, int n_in,
                              void* d_out, int out_size, void* d_ws, size_t ws_size,
                              hipStream_t stream) {
  const float* inputs = (const float*)d_in[0];
  const float* hx     = (const float*)d_in[1];
  const float* adj    = (const float*)d_in[2];
  const float* W_ru   = (const float*)d_in[3];
  const float* b_ru   = (const float*)d_in[4];
  const float* W_c    = (const float*)d_in[5];
  const float* b_c    = (const float*)d_in[6];
  float* out = (float*)d_out;
  (void)in_sizes; (void)n_in; (void)out_size; (void)ws_size;

  char* ws = (char*)d_ws;
  size_t off = 0;
  auto take = [&](size_t bytes) -> char* {
    char* p = ws + off;
    off += (bytes + 255) & ~(size_t)255;
    return p;
  };
  float*  invrow = (float*)take((size_t)NB * sizeof(float));
  float*  invcol = (float*)take((size_t)NB * sizeof(float));
  __bf16* S0     = (__bf16*)take((size_t)NB * NB * 2);
  __bf16* S1     = (__bf16*)take((size_t)NB * NB * 2);
  __bf16* X0     = (__bf16*)take((size_t)NB * COLS * 2);
  __bf16* Xa     = (__bf16*)take((size_t)NB * COLS * 2);
  __bf16* Xb     = (__bf16*)take((size_t)NB * COLS * 2);
  __bf16* Wru    = (__bf16*)take((size_t)FD * MMAT * (2 * UD) * 2);  // [640 x 128]
  __bf16* Wc     = (__bf16*)take((size_t)FD * MMAT * UD * 2);        // [640 x 64]
  float*  ACCru  = (float*)take((size_t)RR * FD * sizeof(float));
  float*  ACCc   = (float*)take((size_t)RR * UD * sizeof(float));

  // Supports + operand precision conversion
  k_degree  <<<NB, 256, 0, stream>>>(adj, invrow, invcol);
  k_supports<<<(NB * NB) / 256, 256, 0, stream>>>(adj, invrow, invcol, S0, S1);
  k_cast_bf16<<<(FD * MMAT * 2 * UD + 255) / 256, 256, 0, stream>>>(W_ru, Wru, FD * MMAT * 2 * UD);
  k_cast_bf16<<<(FD * MMAT * UD + 255) / 256, 256, 0, stream>>>(W_c, Wc, FD * MMAT * UD);
  k_concat<<<(int)(((size_t)NB * COLS) / 256), 256, 0, stream>>>(inputs, hx, X0);

  // diffusion step (wide tiles): Y = S@X (Z=null) or Y = 2*(S@X) - Z
  auto diff = [&](const __bf16* S, const __bf16* X, const __bf16* Z, __bf16* Y) {
    k_wmma_gemm<4><<<dim3(COLS / 128, NB / 64), 128, 0, stream>>>(
        S, NB, X, COLS, Z, Y, nullptr, /*K=*/NB, /*ldc=*/COLS, Z ? 1 : 0);
  };
  // projection accumulate (narrow tiles): ACC += Xm[R x 128] @ W(f*MMAT+m, :P)
  auto proj = [&](const __bf16* Xm, const __bf16* W, int m, int P, float* ACC) {
    k_wmma_gemm<2><<<dim3(P / 64, RR / 64), 128, 0, stream>>>(
        Xm, FD, W + m * P, MMAT * P, nullptr, nullptr, ACC, /*K=*/FD, /*ldc=*/P, 2);
  };

  // ---- gconv 1: r,u gates ----
  k_init_acc<<<(int)(((size_t)RR * FD) / 256), 256, 0, stream>>>(ACCru, b_ru, FD);
  proj(X0, Wru, 0, FD, ACCru);
  diff(S0, X0, nullptr, Xa);  proj(Xa, Wru, 1, FD, ACCru);
  diff(S0, Xa, X0, Xb);       proj(Xb, Wru, 2, FD, ACCru);
  diff(S1, X0, nullptr, Xa);  proj(Xa, Wru, 3, FD, ACCru);
  diff(S1, Xa, X0, Xb);       proj(Xb, Wru, 4, FD, ACCru);
  k_act_gate<<<(int)(((size_t)RR * UD) / 256), 256, 0, stream>>>(ACCru, hx, X0);

  // ---- gconv 2: candidate c (X0 hidden half now holds r*h) ----
  k_init_acc<<<(int)(((size_t)RR * UD) / 256), 256, 0, stream>>>(ACCc, b_c, UD);
  proj(X0, Wc, 0, UD, ACCc);
  diff(S0, X0, nullptr, Xa);  proj(Xa, Wc, 1, UD, ACCc);
  diff(S0, Xa, X0, Xb);       proj(Xb, Wc, 2, UD, ACCc);
  diff(S1, X0, nullptr, Xa);  proj(Xa, Wc, 3, UD, ACCc);
  diff(S1, Xa, X0, Xb);       proj(Xb, Wc, 4, UD, ACCc);

  k_final<<<(int)(((size_t)RR * UD) / 256), 256, 0, stream>>>(ACCru, ACCc, hx, out);
}
